// FAXModule_SwissCheese_661424963681
// MI455X (gfx1250) — matrix operand order repaired and checked by reading, not compiled, not run on hardware
//
#include <hip/hip_runtime.h>

// ---------------------------------------------------------------------------
// CDNA5 (gfx1250) fused windowed attention, bf16 WMMA + f32 accumulate.
// wave32; one wave per workgroup for all WMMA kernels (EXEC all-ones, no
// barriers needed: LDS ops within a wave are in-order).
// GEMMs: 16x64 register blocking (4 accumulators) + explicit double-buffered
// K-loop so fragment loads for chunk k+1 are in flight behind chunk k's WMMAs.
// ---------------------------------------------------------------------------

typedef __bf16 v16bf __attribute__((ext_vector_type(16)));
typedef float  v8f   __attribute__((ext_vector_type(8)));

union FragBF { v16bf v; uint4 q[2]; };

constexpr int NTOK   = 625;   // 25*25 tokens
constexpr int NPAD   = 640;   // padded to 40 tiles of 16
constexpr int DMODEL = 512;
constexpr int NHEADS = 8;
constexpr int DHEAD  = 64;
constexpr int BATCH  = 64;
constexpr int D3     = 3 * DMODEL;

// workspace layout (bytes)
constexpr size_t OFF_WQKVT = 0;
constexpr size_t SZ_WQKVT  = (size_t)D3 * DMODEL * 2;                 // 1.5 MB
constexpr size_t OFF_WOUTT = OFF_WQKVT + SZ_WQKVT;
constexpr size_t SZ_WOUTT  = (size_t)DMODEL * DMODEL * 2;             // 0.5 MB
constexpr size_t OFF_BIAS  = OFF_WOUTT + SZ_WOUTT;
constexpr size_t SZ_BIAS   = (size_t)NHEADS * NPAD * NPAD * 4;        // 13.1 MB
constexpr size_t OFF_X     = OFF_BIAS + SZ_BIAS;
constexpr size_t SZ_X      = (size_t)BATCH * NPAD * DMODEL * 2;       // 41.9 MB
constexpr size_t OFF_AO    = OFF_X;                                   // alias: X dead after QKV GEMM
constexpr size_t OFF_Q     = OFF_X + SZ_X;
constexpr size_t SZ_H      = (size_t)BATCH * NHEADS * NPAD * DHEAD * 2; // 41.9 MB
constexpr size_t OFF_K     = OFF_Q + SZ_H;
constexpr size_t OFF_V     = OFF_K + SZ_H;                            // total ~183 MB

// ---------------------------------------------------------------------------
// WMMA fragment loaders (contiguous k = 16h + s, same order in A and B), bf16, row-major sources.
// A (16x32): lane l -> row = l&15; half h=l>>4:
//   VGPR0-7 = K (h*16 .. h*16+15)
// B (32x16): lane l -> col N = l&15; half h: VGPR0-7 = K (h*16 .. h*16+15)
// C/D (16x16 f32): VGPR g, lanes 0-15 -> (row g, col lane); lanes 16-31 -> row g+8
// ---------------------------------------------------------------------------

__device__ __forceinline__ v16bf load_frag_A(const __bf16* base, int stride,
                                             int row0, int k0) {
  const int lane = threadIdx.x & 31;
  const int r = lane & 15, h = lane >> 4;
  const __bf16* p = base + (size_t)(row0 + r) * stride + k0 + h * 16;
  FragBF f;
  f.q[0] = *(const uint4*)p;        // K = h*16 .. h*16+7
  f.q[1] = *(const uint4*)(p + 8);  // K = h*16+8 .. h*16+15
  return f.v;
}

__device__ __forceinline__ v16bf load_frag_B(const __bf16* base, int stride,
                                             int col0, int k0) {
  const int lane = threadIdx.x & 31;
  const int n = lane & 15, h = lane >> 4;
  const __bf16* p = base + (size_t)(col0 + n) * stride + k0 + h * 16;
  FragBF f;
  f.q[0] = *(const uint4*)p;        // K = h*16 .. h*16+7
  f.q[1] = *(const uint4*)(p + 8);  // K = h*16+8 .. h*16+15
  return f.v;
}

__device__ __forceinline__ v8f wmma_bf16(v16bf a, v16bf b, v8f c) {
  return __builtin_amdgcn_wmma_f32_16x16x32_bf16(false, a, false, b,
                                                 (short)0, c, false, false);
}

// ---------------------------------------------------------------------------
// K0a: weights -> K-contiguous bf16 transposes
// ---------------------------------------------------------------------------
__global__ void prep_weights(const float* __restrict__ w_qkv,
                             const float* __restrict__ w_out,
                             __bf16* __restrict__ WqkvT,
                             __bf16* __restrict__ WoutT) {
  const int i = blockIdx.x * 256 + threadIdx.x;   // 786432 = 1536*512
  {
    const int j = i / DMODEL, k = i % DMODEL;     // WqkvT[j][k] = w_qkv[k][j]
    WqkvT[i] = (__bf16)w_qkv[(size_t)k * D3 + j];
  }
  if (i < DMODEL * DMODEL) {
    const int j = i / DMODEL, k = i % DMODEL;
    WoutT[i] = (__bf16)w_out[(size_t)k * DMODEL + j];
  }
}

// ---------------------------------------------------------------------------
// K0b: Bias[m][i][j] = rel_pos_bias[rel_pos_indices[i,j], m]  (padded, f32)
// ---------------------------------------------------------------------------
__global__ void prep_bias(const float* __restrict__ rpb,
                          const int* __restrict__ rpi,
                          float* __restrict__ Bias) {
  const int i = blockIdx.x * 256 + threadIdx.x;   // 409600 = 640*640
  const int jj = i % NPAD, ii = i / NPAD;
  float vals[NHEADS];
  if (ii < NTOK && jj < NTOK) {
    const int idx = rpi[ii * NTOK + jj];
#pragma unroll
    for (int m = 0; m < NHEADS; ++m) vals[m] = rpb[idx * NHEADS + m];
  } else {
#pragma unroll
    for (int m = 0; m < NHEADS; ++m) vals[m] = 0.f;
  }
#pragma unroll
  for (int m = 0; m < NHEADS; ++m)
    Bias[((size_t)m * NPAD + ii) * NPAD + jj] = vals[m];
}

// ---------------------------------------------------------------------------
// K1: x (b,d,n) f32 -> tokens Xbf (b, NPAD, d) bf16, zero-padded rows
// ---------------------------------------------------------------------------
__global__ void x_to_tokens(const float* __restrict__ x,
                            __bf16* __restrict__ X) {
  const size_t i = (size_t)blockIdx.x * 256 + threadIdx.x; // 64*640*512
  const int dd = (int)(i & (DMODEL - 1));
  const int n  = (int)((i >> 9) % NPAD);
  const size_t b = i / ((size_t)NPAD * DMODEL);
  const float v = (n < NTOK) ? x[((size_t)b * DMODEL + dd) * NTOK + n] : 0.f;
  X[i] = (__bf16)v;
}

// ---------------------------------------------------------------------------
// Double-buffered 16x64 GEMM core: C[4] (16x16 each) += A(16x512) * B(512x64)
// ---------------------------------------------------------------------------
__device__ __forceinline__ void gemm_16x64(const __bf16* __restrict__ A, int i0,
                                           const __bf16* __restrict__ B, int j0,
                                           v8f acc[4]) {
  v16bf a  = load_frag_A(A, DMODEL, i0, 0);
  v16bf b0 = load_frag_B(B, DMODEL, j0 + 0,  0);
  v16bf b1 = load_frag_B(B, DMODEL, j0 + 16, 0);
  v16bf b2 = load_frag_B(B, DMODEL, j0 + 32, 0);
  v16bf b3 = load_frag_B(B, DMODEL, j0 + 48, 0);
#pragma unroll
  for (int k0 = 0; k0 < DMODEL; k0 += 32) {
    v16bf an, bn0, bn1, bn2, bn3;
    if (k0 + 32 < DMODEL) {           // issue next chunk's loads first
      __builtin_prefetch(A + (size_t)i0 * DMODEL + k0 + 96, 0, 1);
      an  = load_frag_A(A, DMODEL, i0, k0 + 32);
      bn0 = load_frag_B(B, DMODEL, j0 + 0,  k0 + 32);
      bn1 = load_frag_B(B, DMODEL, j0 + 16, k0 + 32);
      bn2 = load_frag_B(B, DMODEL, j0 + 32, k0 + 32);
      bn3 = load_frag_B(B, DMODEL, j0 + 48, k0 + 32);
    }
    acc[0] = wmma_bf16(a, b0, acc[0]);
    acc[1] = wmma_bf16(a, b1, acc[1]);
    acc[2] = wmma_bf16(a, b2, acc[2]);
    acc[3] = wmma_bf16(a, b3, acc[3]);
    a = an; b0 = bn0; b1 = bn1; b2 = bn2; b3 = bn3;
  }
}

// ---------------------------------------------------------------------------
// K2: QKV GEMM.  One wave per 16x64 tile; 64 bf16 WMMAs per wave.
// A 64-wide tile never crosses a q/k/v section or head boundary.
// q scaled by dh^-0.5; K row-major [n][dh]; V stored transposed [dh][n].
// ---------------------------------------------------------------------------
__global__ void __launch_bounds__(32, 1)
qkv_gemm(const __bf16* __restrict__ X, const __bf16* __restrict__ WqkvT,
         __bf16* __restrict__ Qb, __bf16* __restrict__ Kb,
         __bf16* __restrict__ Vt) {
  const int i0 = blockIdx.x * 16;       // token tile
  const int j0 = blockIdx.y * 64;       // output-channel tile (0..1535, 64-wide)
  const int b  = blockIdx.z;
  const __bf16* A = X + (size_t)b * NPAD * DMODEL;

  v8f acc[4] = {};
  gemm_16x64(A, i0, WqkvT, j0, acc);

  const int lane = threadIdx.x & 31;
  const int col = lane & 15, h = lane >> 4;
  const int sec = j0 >> 9;              // 0=q 1=k 2=v (tile within one section)
  const int m   = (j0 & 511) >> 6;      // head (tile within one head)
  const size_t hb = (size_t)(b * NHEADS + m);

  if (sec == 0) {
    __bf16* dst = Qb + hb * NPAD * DHEAD;
#pragma unroll
    for (int t = 0; t < 4; ++t)
#pragma unroll
      for (int g = 0; g < 8; ++g)
        dst[(size_t)(i0 + g + 8 * h) * DHEAD + t * 16 + col] =
            (__bf16)(acc[t][g] * 0.125f);
  } else if (sec == 1) {
    __bf16* dst = Kb + hb * NPAD * DHEAD;
#pragma unroll
    for (int t = 0; t < 4; ++t)
#pragma unroll
      for (int g = 0; g < 8; ++g)
        dst[(size_t)(i0 + g + 8 * h) * DHEAD + t * 16 + col] = (__bf16)acc[t][g];
  } else {
#pragma unroll
    for (int t = 0; t < 4; ++t) {
      __bf16* dst = Vt + (hb * DHEAD + t * 16 + col) * NPAD;
#pragma unroll
      for (int g = 0; g < 8; ++g)
        dst[i0 + g + 8 * h] = (__bf16)acc[t][g];
    }
  }
}

// ---------------------------------------------------------------------------
// K3: flash attention.  One wave per (b, head, 16-row tile); stream 32 cols
// per step: 4 WMMAs for S = q@k^T, online softmax (cross-lane max/sum within
// each 16-lane half via shfl_xor), P restaged through LDS (C-layout ->
// A-layout), 4 WMMAs for O += P@V.
// ---------------------------------------------------------------------------
__global__ void __launch_bounds__(32, 1)
attn_kernel(const __bf16* __restrict__ Qb, const __bf16* __restrict__ Kb,
            const __bf16* __restrict__ Vt, const float* __restrict__ Bias,
            __bf16* __restrict__ AO) {
  __shared__ __align__(16) __bf16 Pl[16 * 32];

  const int i0 = blockIdx.x * 16;
  const int m  = blockIdx.y;
  const int b  = blockIdx.z;
  const int lane = threadIdx.x & 31;
  const int col = lane & 15, h = lane >> 4;

  const size_t hb = (size_t)(b * NHEADS + m);
  const __bf16* Qh = Qb + hb * NPAD * DHEAD;
  const __bf16* Kh = Kb + hb * NPAD * DHEAD;
  const __bf16* Vh = Vt + hb * DHEAD * NPAD;
  const float*  Bm = Bias + (size_t)m * NPAD * NPAD;

  const v16bf qa0 = load_frag_A(Qh, DHEAD, i0, 0);
  const v16bf qa1 = load_frag_A(Qh, DHEAD, i0, 32);

  v8f o0 = {}, o1 = {}, o2 = {}, o3 = {};
  float mrow[8], lsum[8];
#pragma unroll
  for (int g = 0; g < 8; ++g) { mrow[g] = -3.0e38f; lsum[g] = 0.f; }

  for (int jt = 0; jt < NPAD; jt += 32) {
    // ---- S = q @ k^T for two 16-col subtiles (loads hoisted) -------------
    const v16bf kb00 = load_frag_B(Kh, DHEAD, jt, 0);
    const v16bf kb01 = load_frag_B(Kh, DHEAD, jt, 32);
    const v16bf kb10 = load_frag_B(Kh, DHEAD, jt + 16, 0);
    const v16bf kb11 = load_frag_B(Kh, DHEAD, jt + 16, 32);
    v8f s0 = {}, s1 = {};
    s0 = wmma_bf16(qa0, kb00, s0);
    s0 = wmma_bf16(qa1, kb01, s0);
    s1 = wmma_bf16(qa0, kb10, s1);
    s1 = wmma_bf16(qa1, kb11, s1);
    // ---- bias + mask ------------------------------------------------------
    const int jA = jt + col, jB = jt + 16 + col;
    const float* bp = Bm + (size_t)(i0 + 8 * h) * NPAD;
#pragma unroll
    for (int g = 0; g < 8; ++g) {
      const float bsA = s0[g] + bp[(size_t)g * NPAD + jA];
      const float bsB = s1[g] + bp[(size_t)g * NPAD + jB];
      s0[g] = (jA < NTOK) ? bsA : -1.0e30f;
      s1[g] = (jB < NTOK) ? bsB : -1.0e30f;
    }
    // ---- online softmax update (rows live in one 16-lane half) ----------
#pragma unroll
    for (int g = 0; g < 8; ++g) {
      float tm = fmaxf(s0[g], s1[g]);
      tm = fmaxf(tm, __shfl_xor(tm, 8, 32));
      tm = fmaxf(tm, __shfl_xor(tm, 4, 32));
      tm = fmaxf(tm, __shfl_xor(tm, 2, 32));
      tm = fmaxf(tm, __shfl_xor(tm, 1, 32));
      const float mn = fmaxf(mrow[g], tm);
      const float sc = __expf(mrow[g] - mn);
      mrow[g] = mn;
      const float p0 = __expf(s0[g] - mn);
      const float p1 = __expf(s1[g] - mn);
      lsum[g] = lsum[g] * sc + p0 + p1;             // per-lane partial sum
      o0[g] *= sc; o1[g] *= sc; o2[g] *= sc; o3[g] *= sc;
      Pl[(g + 8 * h) * 32 + col]      = (__bf16)p0; // C-layout -> LDS
      Pl[(g + 8 * h) * 32 + 16 + col] = (__bf16)p1;
    }
    // ---- O += P @ V  (reload P in A-layout; wave-in-order DS) ------------
    const v16bf vb0 = load_frag_B(Vh, NPAD, 0,  jt);
    const v16bf vb1 = load_frag_B(Vh, NPAD, 16, jt);
    const v16bf vb2 = load_frag_B(Vh, NPAD, 32, jt);
    const v16bf vb3 = load_frag_B(Vh, NPAD, 48, jt);
    const v16bf pa = load_frag_A(Pl, 32, 0, 0);
    o0 = wmma_bf16(pa, vb0, o0);
    o1 = wmma_bf16(pa, vb1, o1);
    o2 = wmma_bf16(pa, vb2, o2);
    o3 = wmma_bf16(pa, vb3, o3);
  }

  // ---- finalize: reduce row sums across the half, normalize, store ------
#pragma unroll
  for (int g = 0; g < 8; ++g) {
    float v = lsum[g];
    v += __shfl_xor(v, 8, 32);
    v += __shfl_xor(v, 4, 32);
    v += __shfl_xor(v, 2, 32);
    v += __shfl_xor(v, 1, 32);
    lsum[g] = __frcp_rn(v);
  }
  __bf16* dst = AO + ((size_t)b * NPAD + i0) * DMODEL + m * DHEAD;
#pragma unroll
  for (int g = 0; g < 8; ++g) {
    const size_t ro = (size_t)(g + 8 * h) * DMODEL;
    dst[ro + 0  + col] = (__bf16)(o0[g] * lsum[g]);
    dst[ro + 16 + col] = (__bf16)(o1[g] * lsum[g]);
    dst[ro + 32 + col] = (__bf16)(o2[g] * lsum[g]);
    dst[ro + 48 + col] = (__bf16)(o3[g] * lsum[g]);
  }
}

// ---------------------------------------------------------------------------
// K4: out = (AO @ w_out)^T.  One wave per 16x64 tile; 64 bf16 WMMAs per
// wave, then LDS transpose so the final (b, d, n) f32 store is
// 64B-contiguous per output row.
// ---------------------------------------------------------------------------
__global__ void __launch_bounds__(32, 1)
out_gemm(const __bf16* __restrict__ AO, const __bf16* __restrict__ WoutT,
         float* __restrict__ out) {
  __shared__ __align__(16) float T[64 * 16];   // [d within tile][token]

  const int i0 = blockIdx.x * 16;       // token tile
  const int j0 = blockIdx.y * 64;       // d tile (64-wide)
  const int b  = blockIdx.z;
  const __bf16* A = AO + (size_t)b * NPAD * DMODEL;

  v8f acc[4] = {};
  gemm_16x64(A, i0, WoutT, j0, acc);

  const int lane = threadIdx.x & 31;
  const int col = lane & 15, h = lane >> 4;
#pragma unroll
  for (int t = 0; t < 4; ++t)
#pragma unroll
    for (int g = 0; g < 8; ++g)
      T[(t * 16 + col) * 16 + g + 8 * h] = acc[t][g];  // wave-in-order DS

#pragma unroll
  for (int it = 0; it < 32; ++it) {
    const int flat = it * 32 + lane;    // 1024 elements
    const int cc = flat >> 4, t = flat & 15;
    const int token = i0 + t;
    if (token < NTOK)
      out[((size_t)b * DMODEL + j0 + cc) * NTOK + token] = T[flat];
  }
}

// ---------------------------------------------------------------------------
extern "C" void kernel_launch(void* const* d_in, const int* in_sizes, int n_in,
                              void* d_out, int out_size, void* d_ws,
                              size_t ws_size, hipStream_t stream) {
  const float* x     = (const float*)d_in[0];
  const float* w_qkv = (const float*)d_in[1];
  const float* w_out = (const float*)d_in[2];
  const float* rpb   = (const float*)d_in[3];
  const int*   rpi   = (const int*)d_in[4];
  float* out = (float*)d_out;

  char* ws = (char*)d_ws;
  __bf16* WqkvT = (__bf16*)(ws + OFF_WQKVT);
  __bf16* WoutT = (__bf16*)(ws + OFF_WOUTT);
  float*  Bias  = (float*)(ws + OFF_BIAS);
  __bf16* Xbf   = (__bf16*)(ws + OFF_X);
  __bf16* AO    = (__bf16*)(ws + OFF_AO);   // aliases Xbf (X dead after K2)
  __bf16* Qb    = (__bf16*)(ws + OFF_Q);
  __bf16* Kb    = (__bf16*)(ws + OFF_K);
  __bf16* Vt    = (__bf16*)(ws + OFF_V);

  prep_weights<<<(D3 * DMODEL) / 256, 256, 0, stream>>>(w_qkv, w_out, WqkvT, WoutT);
  prep_bias<<<(NPAD * NPAD) / 256, 256, 0, stream>>>(rpb, rpi, Bias);
  x_to_tokens<<<(BATCH * NPAD * DMODEL) / 256, 256, 0, stream>>>(x, Xbf);
  qkv_gemm<<<dim3(NPAD / 16, D3 / 64, BATCH), 32, 0, stream>>>(Xbf, WqkvT, Qb, Kb, Vt);
  attn_kernel<<<dim3(NPAD / 16, NHEADS, BATCH), 32, 0, stream>>>(Qb, Kb, Vt, Bias, AO);
  out_gemm<<<dim3(NPAD / 16, DMODEL / 64, BATCH), 32, 0, stream>>>(AO, WoutT, out);
}
